// TransformerEncoderLayer_MOE_66022237274327
// MI455X (gfx1250) — matrix operand order repaired and checked by reading, not compiled, not run on hardware
//
#include <hip/hip_runtime.h>

#define S_  1024
#define B_  8
#define D_  1024
#define H_  16
#define HD_ 64
#define F_  4096
#define E_  8
#define N_  (S_ * B_)          // 8192 tokens
#define CAP_ (N_ / E_)         // 1024 slots per expert

typedef __attribute__((ext_vector_type(16))) __bf16 v16bf;
typedef __attribute__((ext_vector_type(8)))  float  v8f;

// ---------------- bf16 helpers (stored as raw u16 bits) ----------------
static __device__ __forceinline__ unsigned short f2bf(float f) {
    union { float f; unsigned u; } c; c.f = f;
    unsigned r = c.u + 0x7FFFu + ((c.u >> 16) & 1u);   // round-to-nearest-even
    return (unsigned short)(r >> 16);
}

// ---------------- WMMA fragment loaders (ISA 7.12.2 layouts, wave32) ----
union FragCvt { uint4 q[2]; v16bf v; };
static __device__ __forceinline__ v16bf ld_frag(const unsigned short* lo,
                                                const unsigned short* hi) {
    FragCvt u;
    u.q[0] = *(const uint4*)lo;
    u.q[1] = *(const uint4*)hi;
    return u.v;
}
// A 16x32 bf16, row-major [M][K] in LDS: lane<16 row=lane K={0..7,16..23}; hi lanes K+8
static __device__ __forceinline__ v16bf load_a(const unsigned short* sm, int stride,
                                               int m_base, int k_base, int lane) {
    const unsigned short* p = sm + (m_base + (lane & 15)) * stride + k_base + ((lane & 16) ? 8 : 0);
    return ld_frag(p, p + 16);
}
// B 32x16 bf16 stored N-major [N][K] in LDS: lane&15 = N; lane<16 K={0..7,16..23}; hi lanes K+8
static __device__ __forceinline__ v16bf load_b(const unsigned short* sm, int stride,
                                               int k_base, int n_base, int lane) {
    const unsigned short* p = sm + (n_base + (lane & 15)) * stride + k_base + ((lane & 16) ? 8 : 0);
    return ld_frag(p, p + 16);
}

// ---------------- elementwise utility kernels ----------------
__global__ void f32_to_bf16_kernel(const float* __restrict__ s,
                                   unsigned short* __restrict__ d, long long n) {
    long long i = (long long)blockIdx.x * 256 + threadIdx.x;
    if (i < n) d[i] = f2bf(s[i]);
}
__global__ void zero_u16_kernel(unsigned short* __restrict__ p, long long n) {
    long long i = (long long)blockIdx.x * 256 + threadIdx.x;
    if (i < n) p[i] = 0;
}
__global__ void zero_f32_kernel(float* __restrict__ p, int n) {
    int i = blockIdx.x * 256 + threadIdx.x;
    if (i < n) p[i] = 0.f;
}

// ---------------- transpose + convert: src f32 [R][C] -> dst bf16 [C][R] ----
__global__ __launch_bounds__(256) void tcvt_kernel(const float* __restrict__ src,
                                                   unsigned short* __restrict__ dst,
                                                   int R, int C, long long sS, long long sD) {
    __shared__ float t[32][33];
    int z = blockIdx.z;
    src += (long long)z * sS;
    dst += (long long)z * sD;
    int r0 = blockIdx.y * 32, c0 = blockIdx.x * 32;
    int tx = threadIdx.x & 31, ty = threadIdx.x >> 5;   // 8 rows of 32
#pragma unroll
    for (int i = 0; i < 32; i += 8)
        t[ty + i][tx] = src[(long long)(r0 + ty + i) * C + c0 + tx];
    __syncthreads();
#pragma unroll
    for (int i = 0; i < 32; i += 8)
        dst[(long long)(c0 + ty + i) * R + r0 + tx] = f2bf(t[tx][ty + i]);
}

// ---------------- LayerNorm: one block per token row ----------------
__global__ __launch_bounds__(256) void ln_kernel(const float* __restrict__ x,
                                                 const float* __restrict__ g,
                                                 const float* __restrict__ bt,
                                                 float* __restrict__ outF,
                                                 unsigned short* __restrict__ outB) {
    __shared__ float red[256];
    int n = blockIdx.x, tid = threadIdx.x;
    const float* row = x + (long long)n * D_;
    float s = 0.f;
#pragma unroll
    for (int i = 0; i < D_ / 256; ++i) s += row[tid + i * 256];
    red[tid] = s; __syncthreads();
    for (int o = 128; o > 0; o >>= 1) { if (tid < o) red[tid] += red[tid + o]; __syncthreads(); }
    float mean = red[0] / (float)D_;
    __syncthreads();
    float v = 0.f;
#pragma unroll
    for (int i = 0; i < D_ / 256; ++i) { float c = row[tid + i * 256] - mean; v += c * c; }
    red[tid] = v; __syncthreads();
    for (int o = 128; o > 0; o >>= 1) { if (tid < o) red[tid] += red[tid + o]; __syncthreads(); }
    float rstd = rsqrtf(red[0] / (float)D_ + 1e-5f);
#pragma unroll
    for (int i = 0; i < D_ / 256; ++i) {
        int d = tid + i * 256;
        float val = (row[d] - mean) * rstd * g[d] + bt[d];
        if (outF) outF[(long long)n * D_ + d] = val;
        outB[(long long)n * D_ + d] = f2bf(val);
    }
}

// ---------------- generic bf16 WMMA GEMM ----------------
// out[M,N] = relu?( alpha*(A@Wt^T + bias) ) + res
// A: [M,K] bf16 row-major;  Wt: [N,K] bf16 (pre-transposed weights).
// Block tile 64(M) x 128(N); 4 waves, each wave 16(M) x 128(N).
// M % 64 == 0, N % 128 == 0, K % 32 == 0 (guaranteed by caller).
__global__ __launch_bounds__(128) void gemm_bf16_kernel(
    const unsigned short* __restrict__ A, const unsigned short* __restrict__ Wt,
    const float* __restrict__ bias, const float* __restrict__ res,
    float* __restrict__ outF, unsigned short* __restrict__ outB,
    int M, int N, int K, float alpha, int relu,
    long long sA, long long sW, long long sBias, long long sOut) {
    __shared__ unsigned short As[64][40];    // 64 rows x 32 K  (+pad)
    __shared__ unsigned short Ws[128][40];   // 128 N-rows x 32 K (+pad)
    int z = blockIdx.z;
    A  += (long long)z * sA;
    Wt += (long long)z * sW;
    const float* biasz = bias ? bias + (long long)z * sBias : nullptr;
    long long ob = (long long)z * sOut;

    int m0 = blockIdx.y * 64, n0 = blockIdx.x * 128;
    int tid = threadIdx.x, lane = tid & 31, wm = tid >> 5;
    v8f acc[8] = {};

    const int ra = tid >> 1, ca = (tid & 1) * 16;   // A stage mapping
    for (int k0 = 0; k0 < K; k0 += 32) {
        // stage A tile 64x32 (2 x uint4 per thread, fully vectorized)
        {
            const uint4* srcA = (const uint4*)(A + (long long)(m0 + ra) * K + k0 + ca);
            __builtin_prefetch(srcA + 4, 0, 1);
            uint4* dA = (uint4*)&As[ra][ca];
            dA[0] = srcA[0]; dA[1] = srcA[1];
            // stage Wt tile 128x32 (one N-row per thread, 4 x uint4)
            const uint4* srcW = (const uint4*)(Wt + (long long)(n0 + tid) * K + k0);
            __builtin_prefetch(srcW + 8, 0, 1);
            uint4* dW = (uint4*)&Ws[tid][0];
            dW[0] = srcW[0]; dW[1] = srcW[1]; dW[2] = srcW[2]; dW[3] = srcW[3];
        }
        __syncthreads();
        v16bf a = load_a(&As[0][0], 40, wm * 16, 0, lane);
#pragma unroll
        for (int j = 0; j < 8; ++j) {
            v16bf b = load_b(&Ws[0][0], 40, 0, j * 16, lane);
            acc[j] = __builtin_amdgcn_wmma_f32_16x16x32_bf16(
                false, a, false, b, (short)0, acc[j], false, false);
        }
        __syncthreads();
    }
    // epilogue (C/D layout: VGPR r -> M = r + 8*(lane>=16), N = lane&15)
    int hi8 = (lane & 16) ? 8 : 0;
#pragma unroll
    for (int j = 0; j < 8; ++j) {
        int gn = n0 + j * 16 + (lane & 15);
        float bv = biasz ? biasz[gn] : 0.f;
#pragma unroll
        for (int r = 0; r < 8; ++r) {
            int gm = m0 + wm * 16 + r + hi8;
            float v = (acc[j][r] + bv) * alpha;
            if (relu) v = fmaxf(v, 0.f);
            if (res) v += res[(long long)gm * N + gn];
            long long oi = ob + (long long)gm * N + gn;
            if (outF) outF[oi] = v;
            if (outB) outB[oi] = f2bf(v);
        }
    }
}

// ---------------- flash attention (bf16 WMMA, online softmax) ----------------
// q/k/v: [N,D] bf16 token-major (token = s*B+b, col = h*HD+d); q pre-scaled.
__global__ __launch_bounds__(128) void attn_fa_kernel(
    const unsigned short* __restrict__ qb, const unsigned short* __restrict__ kb,
    const unsigned short* __restrict__ vb, const unsigned char* __restrict__ mask,
    unsigned short* __restrict__ attnb) {
    __shared__ unsigned short Qs[64][72];   // [q row][d]      (A of Q@K^T)
    __shared__ unsigned short Ks[64][72];   // [k col][d]      (B of Q@K^T, N-major)
    __shared__ unsigned short Vt[64][72];   // [d][k row]      (B of P@V,  N-major)
    __shared__ unsigned short Ps[4][16][72];// per-wave P tile (A of P@V)
    int bh = blockIdx.y;
    int b = bh >> 4, h = bh & 15;
    int q0 = blockIdx.x * 64;
    int tid = threadIdx.x, lane = tid & 31, w = tid >> 5;
    int hi8 = (lane & 16) ? 8 : 0;
    const int rr = tid >> 1, cc = (tid & 1) * 32;

    // stage Q block 64x64 (vectorized)
    {
        const uint4* src = (const uint4*)(qb + ((long long)(q0 + rr) * B_ + b) * D_ + h * HD_ + cc);
        uint4* dq = (uint4*)&Qs[rr][cc];
#pragma unroll
        for (int i = 0; i < 4; ++i) dq[i] = src[i];
    }
    __syncthreads();
    v16bf aq0 = load_a(&Qs[0][0], 72, w * 16, 0, lane);
    v16bf aq1 = load_a(&Qs[0][0], 72, w * 16, 32, lane);

    v8f o[4] = {};
    float rm[8], rl[8];
#pragma unroll
    for (int r = 0; r < 8; ++r) { rm[r] = -1e30f; rl[r] = 0.f; }

    for (int kv0 = 0; kv0 < S_; kv0 += 64) {
        __syncthreads();   // previous iteration's Ks/Vt/Ps reads complete
        {
            long long nn = ((long long)(kv0 + rr) * B_ + b) * D_ + h * HD_ + cc;
            const uint4* ksrc = (const uint4*)(kb + nn);
            uint4* dk = (uint4*)&Ks[rr][cc];
#pragma unroll
            for (int i = 0; i < 4; ++i) dk[i] = ksrc[i];
            // V transposed into Vt[d][kv_local]
            const uint4* vsrc = (const uint4*)(vb + nn);
#pragma unroll
            for (int qq = 0; qq < 4; ++qq) {
                uint4 vv = vsrc[qq];
                unsigned uu[4] = { vv.x, vv.y, vv.z, vv.w };
#pragma unroll
                for (int j = 0; j < 4; ++j) {
                    int d = cc + qq * 8 + 2 * j;
                    Vt[d][rr]     = (unsigned short)(uu[j] & 0xFFFFu);
                    Vt[d + 1][rr] = (unsigned short)(uu[j] >> 16);
                }
            }
        }
        __syncthreads();

        // scores tile 16(q) x 64(k): S = Q @ K^T, HD=64 -> 2 K-steps
        v8f sc[4] = {};
#pragma unroll
        for (int j = 0; j < 4; ++j) {
            v16bf bk = load_b(&Ks[0][0], 72, 0, j * 16, lane);
            sc[j] = __builtin_amdgcn_wmma_f32_16x16x32_bf16(false, aq0, false, bk, (short)0, sc[j], false, false);
            bk = load_b(&Ks[0][0], 72, 32, j * 16, lane);
            sc[j] = __builtin_amdgcn_wmma_f32_16x16x32_bf16(false, aq1, false, bk, (short)0, sc[j], false, false);
        }
        // padding mask
#pragma unroll
        for (int j = 0; j < 4; ++j) {
            int kcol = kv0 + j * 16 + (lane & 15);
            if (mask[b * S_ + kcol]) {
#pragma unroll
                for (int r = 0; r < 8; ++r) sc[j][r] = -1e9f;
            }
        }
        // online softmax per q-row (reduce across the 16-lane N group)
#pragma unroll
        for (int r = 0; r < 8; ++r) {
            float mx = fmaxf(fmaxf(sc[0][r], sc[1][r]), fmaxf(sc[2][r], sc[3][r]));
            for (int mk = 8; mk >= 1; mk >>= 1) mx = fmaxf(mx, __shfl_xor(mx, mk, 32));
            float nm = fmaxf(rm[r], mx);
            float al = __expf(rm[r] - nm);
            rm[r] = nm;
            float ss = 0.f;
#pragma unroll
            for (int j = 0; j < 4; ++j) {
                float p = __expf(sc[j][r] - nm);
                sc[j][r] = p;
                ss += p;
            }
            for (int mk = 8; mk >= 1; mk >>= 1) ss += __shfl_xor(ss, mk, 32);
            rl[r] = rl[r] * al + ss;
#pragma unroll
            for (int j = 0; j < 4; ++j) o[j][r] *= al;
        }
        // write P (bf16) through LDS to obtain A-matrix layout
#pragma unroll
        for (int j = 0; j < 4; ++j)
#pragma unroll
            for (int r = 0; r < 8; ++r)
                Ps[w][r + hi8][j * 16 + (lane & 15)] = f2bf(sc[j][r]);
        __syncthreads();
        // O += P @ V   (K-dim = 64 -> 2 steps; Vt is N-major -> contiguous frags)
        v16bf ap0 = load_a(&Ps[w][0][0], 72, 0, 0, lane);
        v16bf ap1 = load_a(&Ps[w][0][0], 72, 0, 32, lane);
#pragma unroll
        for (int j = 0; j < 4; ++j) {
            v16bf bv0 = load_b(&Vt[0][0], 72, 0, j * 16, lane);
            o[j] = __builtin_amdgcn_wmma_f32_16x16x32_bf16(false, ap0, false, bv0, (short)0, o[j], false, false);
            v16bf bv1 = load_b(&Vt[0][0], 72, 32, j * 16, lane);
            o[j] = __builtin_amdgcn_wmma_f32_16x16x32_bf16(false, ap1, false, bv1, (short)0, o[j], false, false);
        }
    }
    // normalize + store to attn[N,D] bf16
    float inv[8];
#pragma unroll
    for (int r = 0; r < 8; ++r) inv[r] = 1.f / fmaxf(rl[r], 1e-20f);
#pragma unroll
    for (int j = 0; j < 4; ++j)
#pragma unroll
        for (int r = 0; r < 8; ++r) {
            int qr = q0 + w * 16 + r + hi8;
            long long nn = (long long)qr * B_ + b;
            attnb[nn * D_ + h * HD_ + j * 16 + (lane & 15)] = f2bf(o[j][r] * inv[r]);
        }
}

// ---------------- MoE gating: one wave per token ----------------
__global__ __launch_bounds__(256) void gating_kernel(
    const float* __restrict__ h2, const float* __restrict__ wg,
    const unsigned char* __restrict__ mask,
    int* __restrict__ eidx, float* __restrict__ gval, float* __restrict__ gsum) {
    int lane = threadIdx.x & 31, w = threadIdx.x >> 5;
    int n = blockIdx.x * 8 + w;
    float acc[E_] = {};
    const float* row = h2 + (long long)n * D_;
#pragma unroll 4
    for (int i = 0; i < D_ / 32; ++i) {
        int d = i * 32 + lane;
        float xv = row[d];
        const float* wr = wg + (long long)d * E_;
#pragma unroll
        for (int e = 0; e < E_; ++e) acc[e] += xv * wr[e];
    }
#pragma unroll
    for (int e = 0; e < E_; ++e)
        for (int mk = 16; mk >= 1; mk >>= 1) acc[e] += __shfl_xor(acc[e], mk, 32);
    if (lane == 0) {
        float mx = acc[0]; int am = 0;
#pragma unroll
        for (int e = 1; e < E_; ++e) if (acc[e] > mx) { mx = acc[e]; am = e; }
        float p[E_], sum = 0.f;
#pragma unroll
        for (int e = 0; e < E_; ++e) { p[e] = __expf(acc[e] - mx); sum += p[e]; }
        float invs = 1.f / sum;
#pragma unroll
        for (int e = 0; e < E_; ++e) { p[e] *= invs; atomicAdd(&gsum[e], p[e]); }
        int s = n / B_, bb = n % B_;
        int used = mask[bb * S_ + s] ? 0 : 1;
        eidx[n] = used ? am : -1;
        gval[n] = p[am];
    }
}

// ---------------- capacity scan: single block, packed 8x16-bit counters ----
__global__ __launch_bounds__(1024) void scan_kernel(
    const int* __restrict__ eidx, const float* __restrict__ gval,
    int* __restrict__ slot, float* __restrict__ gatew, float* __restrict__ msum) {
    __shared__ unsigned long long s0[1024], s1[1024];
    __shared__ unsigned int base[E_];
    int tid = threadIdx.x;
    if (tid < E_) base[tid] = 0;
    __syncthreads();
    for (int iter = 0; iter < N_ / 1024; ++iter) {
        int n = iter * 1024 + tid;
        int e = eidx[n];
        unsigned long long my0 = 0, my1 = 0;
        if (e >= 0) { if (e < 4) my0 = 1ull << (16 * e); else my1 = 1ull << (16 * (e - 4)); }
        s0[tid] = my0; s1[tid] = my1;
        __syncthreads();
        for (int off = 1; off < 1024; off <<= 1) {
            unsigned long long a0 = 0, a1 = 0;
            if (tid >= off) { a0 = s0[tid - off]; a1 = s1[tid - off]; }
            __syncthreads();
            s0[tid] += a0; s1[tid] += a1;
            __syncthreads();
        }
        int sl = E_ * CAP_; float gw = 0.f;
        if (e >= 0) {
            unsigned long long ex = (e < 4) ? (s0[tid] - my0) : (s1[tid] - my1);
            unsigned loc = base[e] + (unsigned)((ex >> (16 * (e & 3))) & 0xFFFFull);
            if (loc < (unsigned)CAP_) { sl = e * CAP_ + (int)loc; gw = gval[n]; }
        }
        slot[n] = sl; gatew[n] = gw;
        __syncthreads();
        if (tid < E_) {
            unsigned tot = (tid < 4) ? (unsigned)((s0[1023] >> (16 * tid)) & 0xFFFFull)
                                     : (unsigned)((s1[1023] >> (16 * (tid - 4))) & 0xFFFFull);
            base[tid] += tot;
        }
        __syncthreads();
    }
    if (tid < E_) msum[tid] = (float)base[tid];
}

// ---------------- dispatch: copy kept token rows into expert slots ----------
__global__ __launch_bounds__(128) void scatter_kernel(
    const unsigned short* __restrict__ h2b, const int* __restrict__ slot,
    unsigned short* __restrict__ einp) {
    int n = blockIdx.x;
    int sl = slot[n];
    if (sl >= E_ * CAP_) return;
    const uint4* src = (const uint4*)(h2b + (long long)n * D_);
    uint4* dst = (uint4*)(einp + (long long)sl * D_);
    dst[threadIdx.x] = src[threadIdx.x];
}

// ---------------- combine: residual + gated expert output (+ gate_loss) -----
__global__ __launch_bounds__(256) void combine_kernel(
    const float* __restrict__ x1, const float* __restrict__ eout,
    const int* __restrict__ slot, const float* __restrict__ gatew,
    const float* __restrict__ gsum, const float* __restrict__ msum,
    float* __restrict__ out) {
    long long idx = (long long)blockIdx.x * 256 + threadIdx.x;
    int n = (int)(idx >> 10), d = (int)(idx & 1023);
    int sl = slot[n];
    float y = 0.f;
    if (sl < E_ * CAP_) y = eout[(long long)sl * D_ + d] * gatew[n];
    out[idx] = x1[idx] + y;
    if (idx == 0) {
        float loss = 0.f;
#pragma unroll
        for (int e = 0; e < E_; ++e)
            loss += (gsum[e] / (float)N_) * (msum[e] / (float)N_);
        out[(long long)N_ * D_] = loss * (float)E_;
    }
}

// ===================== host-side launch =====================
extern "C" void kernel_launch(void* const* d_in, const int* in_sizes, int n_in,
                              void* d_out, int out_size, void* d_ws, size_t ws_size,
                              hipStream_t stream) {
    const float* x    = (const float*)d_in[0];
    const unsigned char* mask = (const unsigned char*)d_in[1];
    const float* wq = (const float*)d_in[2];  const float* bq = (const float*)d_in[3];
    const float* wk = (const float*)d_in[4];  const float* bk = (const float*)d_in[5];
    const float* wv = (const float*)d_in[6];  const float* bv = (const float*)d_in[7];
    const float* wo = (const float*)d_in[8];  const float* bo = (const float*)d_in[9];
    const float* ln1g = (const float*)d_in[10]; const float* ln1b = (const float*)d_in[11];
    const float* ln2g = (const float*)d_in[12]; const float* ln2b = (const float*)d_in[13];
    const float* wg = (const float*)d_in[14];
    const float* w1 = (const float*)d_in[15]; const float* b1 = (const float*)d_in[16];
    const float* w2 = (const float*)d_in[17]; const float* b2 = (const float*)d_in[18];
    float* out = (float*)d_out;

    char* ws = (char*)d_ws;
    size_t off = 0;
    auto alloc = [&](size_t bytes) -> char* {
        char* p = ws + off;
        off = (off + bytes + 255) & ~(size_t)255;
        return p;
    };
    // weights stored TRANSPOSED: [N][K] bf16
    unsigned short* wqt  = (unsigned short*)alloc((size_t)D_ * D_ * 2);
    unsigned short* wkt  = (unsigned short*)alloc((size_t)D_ * D_ * 2);
    unsigned short* wvt  = (unsigned short*)alloc((size_t)D_ * D_ * 2);
    unsigned short* wot  = (unsigned short*)alloc((size_t)D_ * D_ * 2);
    unsigned short* w1t  = (unsigned short*)alloc((size_t)E_ * D_ * F_ * 2);  // [E][F][D]
    unsigned short* w2t  = (unsigned short*)alloc((size_t)E_ * F_ * D_ * 2);  // [E][D][F]
    unsigned short* h1b  = (unsigned short*)alloc((size_t)N_ * D_ * 2);
    unsigned short* qb   = (unsigned short*)alloc((size_t)N_ * D_ * 2);
    unsigned short* kb   = (unsigned short*)alloc((size_t)N_ * D_ * 2);
    unsigned short* vb   = (unsigned short*)alloc((size_t)N_ * D_ * 2);
    unsigned short* attnb= (unsigned short*)alloc((size_t)N_ * D_ * 2);
    float*          x1   = (float*)alloc((size_t)N_ * D_ * 4);
    float*          h2f  = (float*)alloc((size_t)N_ * D_ * 4);
    unsigned short* h2b  = (unsigned short*)alloc((size_t)N_ * D_ * 2);
    unsigned short* einp = (unsigned short*)alloc((size_t)E_ * CAP_ * D_ * 2);
    unsigned short* hmid = (unsigned short*)alloc((size_t)E_ * CAP_ * F_ * 2);
    float*          eout = (float*)alloc((size_t)E_ * CAP_ * D_ * 4);
    int*            eidx = (int*)alloc((size_t)N_ * 4);
    float*          gval = (float*)alloc((size_t)N_ * 4);
    int*            slot = (int*)alloc((size_t)N_ * 4);
    float*          gatew= (float*)alloc((size_t)N_ * 4);
    float*          gsum = (float*)alloc(8 * 4);
    float*          msum = (float*)alloc(8 * 4);

    auto tcvt = [&](const float* s, unsigned short* d, int R, int C, int batch,
                    long long sS, long long sD) {
        dim3 g(C / 32, R / 32, batch);
        tcvt_kernel<<<g, dim3(256), 0, stream>>>(s, d, R, C, sS, sD);
    };
    auto gemm = [&](const unsigned short* A, const unsigned short* W, const float* bias,
                    const float* res, float* oF, unsigned short* oB,
                    int M, int Nc, int K, float alpha, int relu, int batch,
                    long long sA, long long sW, long long sB, long long sOut) {
        dim3 g(Nc / 128, M / 64, batch);
        gemm_bf16_kernel<<<g, dim3(128), 0, stream>>>(A, W, bias, res, oF, oB,
                                                      M, Nc, K, alpha, relu, sA, sW, sB, sOut);
    };

    // 0) weight transpose-conversions + zeroing
    tcvt(wq, wqt, D_, D_, 1, 0, 0);
    tcvt(wk, wkt, D_, D_, 1, 0, 0);
    tcvt(wv, wvt, D_, D_, 1, 0, 0);
    tcvt(wo, wot, D_, D_, 1, 0, 0);
    tcvt(w1, w1t, D_, F_, E_, (long long)D_ * F_, (long long)D_ * F_);
    tcvt(w2, w2t, F_, D_, E_, (long long)F_ * D_, (long long)F_ * D_);
    {
        long long ne = (long long)E_ * CAP_ * D_;
        zero_u16_kernel<<<dim3((unsigned)((ne + 255) / 256)), dim3(256), 0, stream>>>(einp, ne);
        zero_f32_kernel<<<dim3(1), dim3(64), 0, stream>>>(gsum, 8);
    }

    // 1) LN1 -> h1 (bf16)
    ln_kernel<<<dim3(N_), dim3(256), 0, stream>>>(x, ln1g, ln1b, nullptr, h1b);

    // 2) Q/K/V projections (q folded with HD^-0.5)
    const float qscale = 0.125f;  // 64^-0.5
    gemm(h1b, wqt, bq, nullptr, nullptr, qb, N_, D_, D_, qscale, 0, 1, 0, 0, 0, 0);
    gemm(h1b, wkt, bk, nullptr, nullptr, kb, N_, D_, D_, 1.f, 0, 1, 0, 0, 0, 0);
    gemm(h1b, wvt, bv, nullptr, nullptr, vb, N_, D_, D_, 1.f, 0, 1, 0, 0, 0, 0);

    // 3) attention (flash, per (b,h) stripes of 64 q rows)
    attn_fa_kernel<<<dim3(S_ / 64, B_ * H_), dim3(128), 0, stream>>>(qb, kb, vb, mask, attnb);

    // 4) output projection + residual -> x1 (f32)
    gemm(attnb, wot, bo, x, x1, nullptr, N_, D_, D_, 1.f, 0, 1, 0, 0, 0, 0);

    // 5) LN2 -> h2 (f32 for gating, bf16 for FFN)
    ln_kernel<<<dim3(N_), dim3(256), 0, stream>>>(x1, ln2g, ln2b, h2f, h2b);

    // 6) gating + capacity scan + dispatch
    gating_kernel<<<dim3(N_ / 8), dim3(256), 0, stream>>>(h2f, wg, mask, eidx, gval, gsum);
    scan_kernel<<<dim3(1), dim3(1024), 0, stream>>>(eidx, gval, slot, gatew, msum);
    scatter_kernel<<<dim3(N_), dim3(128), 0, stream>>>(h2b, slot, einp);

    // 7) expert FFN: [E] x (CAP x D @ D x F) relu, then (CAP x F @ F x D)
    gemm(einp, w1t, b1, nullptr, nullptr, hmid, CAP_, F_, D_, 1.f, 1, E_,
         (long long)CAP_ * D_, (long long)D_ * F_, F_, (long long)CAP_ * F_);
    gemm(hmid, w2t, b2, nullptr, eout, nullptr, CAP_, D_, F_, 1.f, 0, E_,
         (long long)CAP_ * F_, (long long)F_ * D_, D_, (long long)CAP_ * D_);

    // 8) combine + gate_loss
    combine_kernel<<<dim3((N_ * (long long)D_) / 256), dim3(256), 0, stream>>>(
        x1, eout, slot, gatew, gsum, msum, out);
}
